// DGCNN_50989851738755
// MI455X (gfx1250) — matrix operand order repaired and checked by reading, not compiled, not run on hardware
//
#include <hip/hip_runtime.h>
#include <hip/hip_bf16.h>
#include <math.h>

// ---------------- problem constants (match reference) ----------------
#define NN   100000      // nodes
#define EE   1600000     // edges
#define FIN  128         // input feature dim
#define HH   32          // hidden channels
#define BB   512         // graphs
#define KK   30          // sort-pool k
#define C1   16
#define C2   32
#define KW2  5
#define DLAT 97          // H*L + 1
#define LDZ  128         // padded leading dim for z / zp (cols 97..127 = 0)
#define T2   11          // (K//2 - KW2 + 1)
#define DENSE (T2 * C2)  // 352

typedef __attribute__((ext_vector_type(16))) _Float16 v16h;
typedef __attribute__((ext_vector_type(8)))  float    v8f;

// ---------------------------------------------------------------------
// B-fragment pre-pack: convert f32 weights (optionally transposed, any
// ldb, with K/N zero-padding) into per-lane WMMA B fragments (16 x f16,
// 32 bytes, contiguous).  Layout: frag[(tn*chunks + ch)*32 + lane][16].
// B frag k order used here (contiguous, same as A): lane col n = lane&15, g = lane>>4,
// element e -> k = ch*32 + g*16 + e.
// ---------------------------------------------------------------------
__global__ void pack_b_frag(const float* __restrict__ Bm, int ldb, int btrans,
                            _Float16* __restrict__ frag,
                            int K, int N, int chunks, int tilesN) {
    int t = blockIdx.x * blockDim.x + threadIdx.x;
    int total = tilesN * chunks * 32;
    if (t >= total) return;
    int lane = t & 31;
    int rest = t >> 5;
    int ch = rest % chunks;
    int tn = rest / chunks;
    int g  = lane >> 4;
    int nc = tn * 16 + (lane & 15);
    _Float16* dst = frag + ((size_t)(tn * chunks + ch) * 32 + lane) * 16;
    int kb = ch * 32;
#pragma unroll
    for (int e = 0; e < 16; ++e) {
        int k = kb + g * 16 + e;
        float v = 0.0f;
        if (k < K && nc < N)
            v = btrans ? Bm[(size_t)nc * ldb + k] : Bm[(size_t)k * ldb + nc];
        dst[e] = (_Float16)v;
    }
}

// ---------------------------------------------------------------------
// GEMM via V_WMMA_F32_16X16X32_F16.  One wave computes NT adjacent 16x16
// N-tiles of one 16-row stripe: the A fragment is loaded once and feeds
// NT WMMAs (halves A traffic for NT=2).  Caller guarantees M % 16 == 0,
// zero-padded K (chunks*32 readable floats per A row), lda % 4 == 0 and
// 16B-aligned A base -> unconditional vectorized loads (4 x b128 A,
// 32B per B fragment).  EXEC stays all-ones through the WMMA loop.
// A frag, this code's contiguous k order (same as B): lane g=lane>>4, m=lane&15;
//   e -> k=g*16+e   (one 16-float run)
// C/D (f32 16x16): lane col n=lane&15; vgpr r -> row m=r+8*g.
// ---------------------------------------------------------------------
template <int NT>
__global__ void wmma_gemm_f16f32(const float* __restrict__ A, int lda,
                                 const _Float16* __restrict__ frag,
                                 float* __restrict__ C, int ldc,
                                 const float* __restrict__ bias, int relu,
                                 int M, int chunks, int N,
                                 int tilesNG, int totalWaves)
{
    int wave = blockIdx.x * (blockDim.x >> 5) + (threadIdx.x >> 5);
    if (wave >= totalWaves) return;               // whole-wave uniform exit
    int lane = threadIdx.x & 31;
    int tm = wave / tilesNG;
    int tg = wave - tm * tilesNG;                 // N tile-group index
    int m0 = tm * 16;

    int g  = lane >> 4;
    int mr = lane & 15;
    int arow = m0 + mr;

    const float* arp = A + (size_t)arow * lda + g * 16;
    const v16h* bf[NT];
#pragma unroll
    for (int j = 0; j < NT; ++j)
        bf[j] = (const v16h*)(frag +
                 ((size_t)(tg * NT + j) * chunks * 32 + lane) * 16);

    v8f acc[NT];
#pragma unroll
    for (int j = 0; j < NT; ++j)
#pragma unroll
        for (int i = 0; i < 8; ++i) acc[j][i] = 0.0f;

    for (int ch = 0; ch < chunks; ++ch) {
        const float4* ap = (const float4*)(arp + ch * 32);
        float4 a0 = ap[0];             // k = g*16 + 0..3
        float4 a1 = ap[1];             // k = g*16 + 4..7
        float4 a2 = ap[2];             // k = g*16 + 8..11
        float4 a3 = ap[3];             // k = g*16 + 12..15
        v16h afrag;
        afrag[0]  = (_Float16)a0.x; afrag[1]  = (_Float16)a0.y;
        afrag[2]  = (_Float16)a0.z; afrag[3]  = (_Float16)a0.w;
        afrag[4]  = (_Float16)a1.x; afrag[5]  = (_Float16)a1.y;
        afrag[6]  = (_Float16)a1.z; afrag[7]  = (_Float16)a1.w;
        afrag[8]  = (_Float16)a2.x; afrag[9]  = (_Float16)a2.y;
        afrag[10] = (_Float16)a2.z; afrag[11] = (_Float16)a2.w;
        afrag[12] = (_Float16)a3.x; afrag[13] = (_Float16)a3.y;
        afrag[14] = (_Float16)a3.z; afrag[15] = (_Float16)a3.w;
#pragma unroll
        for (int j = 0; j < NT; ++j) {
            v16h bfrag = bf[j][ch * 32];   // 32B contiguous, 32B-aligned
            acc[j] = __builtin_amdgcn_wmma_f32_16x16x32_f16(
                       false, afrag, false, bfrag, (short)0, acc[j],
                       false, false);
        }
    }

#pragma unroll
    for (int j = 0; j < NT; ++j) {
        int ncol = (tg * NT + j) * 16 + mr;
#pragma unroll
        for (int r = 0; r < 8; ++r) {
            int mrow = m0 + r + 8 * g;
            if (ncol < N) {
                float v = acc[j][r];
                if (bias) v += bias[ncol];
                if (relu) v = fmaxf(v, 0.0f);
                C[(size_t)mrow * ldc + ncol] = v;
            }
        }
    }
}

// ---------------- small utility kernels ----------------
__global__ void fill_f32(float* p, float v, int n) {
    for (int i = blockIdx.x * blockDim.x + threadIdx.x; i < n;
         i += gridDim.x * blockDim.x) p[i] = v;
}

__global__ void fill_i32(int* p, int v, int n) {
    for (int i = blockIdx.x * blockDim.x + threadIdx.x; i < n;
         i += gridDim.x * blockDim.x) p[i] = v;
}

// deg accumulation: dis starts at 1.0 (self loop), add 1 per in-edge
__global__ void deg_edges(const long long* __restrict__ ei, float* dis) {
    int e = blockIdx.x * blockDim.x + threadIdx.x;
    if (e < EE) atomicAdd(&dis[(int)ei[(size_t)EE + e]], 1.0f);
}

__global__ void rsqrt_inplace(float* dis) {
    int i = blockIdx.x * blockDim.x + threadIdx.x;
    if (i < NN) dis[i] = rsqrtf(dis[i]);
}

// agg[col, f..f+3] += hw[row, f..f+3] * dis[row]*dis[col]  (H=32 path)
// one thread per (edge, 4-feature group): 1 b128 load + 4 L2 atomics
__global__ void edge_agg32(const long long* __restrict__ ei,
                           const float* __restrict__ dis,
                           const float* __restrict__ hw,
                           float* __restrict__ agg) {
    int t = blockIdx.x * blockDim.x + threadIdx.x;
    if (t >= EE * 8) return;
    int e = t >> 3;
    int f = (t & 7) * 4;
    int r = (int)ei[e];
    int c = (int)ei[(size_t)EE + e];
    float nrm = dis[r] * dis[c];
    float4 hv = *(const float4*)(hw + (size_t)r * HH + f);
    float* a = agg + (size_t)c * HH + f;
    atomicAdd(a + 0, hv.x * nrm);
    atomicAdd(a + 1, hv.y * nrm);
    atomicAdd(a + 2, hv.z * nrm);
    atomicAdd(a + 3, hv.w * nrm);
}

// H=1 path (last GCN layer)
__global__ void edge_agg1(const long long* __restrict__ ei,
                          const float* __restrict__ dis,
                          const float* __restrict__ hw,
                          float* __restrict__ agg) {
    int e = blockIdx.x * blockDim.x + threadIdx.x;
    if (e >= EE) return;
    int r = (int)ei[e];
    int c = (int)ei[(size_t)EE + e];
    atomicAdd(&agg[c], hw[r] * dis[r] * dis[c]);
}

// z[i, coloff+f..+3] = tanh(agg + hw/deg + bias), vectorized (H=32 path)
__global__ void gcn_combine32(const float* __restrict__ agg,
                              const float* __restrict__ hw,
                              const float* __restrict__ dis,
                              const float* __restrict__ bias,
                              float* __restrict__ z, int coloff) {
    int t = blockIdx.x * blockDim.x + threadIdx.x;
    if (t >= NN * 8) return;
    int i = t >> 3;
    int f = (t & 7) * 4;
    float d = dis[i];
    float d2 = d * d;
    float4 av = *(const float4*)(agg + (size_t)i * HH + f);
    float4 hv = *(const float4*)(hw + (size_t)i * HH + f);
    float4 bv = *(const float4*)(bias + f);
    float4 o;
    o.x = tanhf(av.x + hv.x * d2 + bv.x);
    o.y = tanhf(av.y + hv.y * d2 + bv.y);
    o.z = tanhf(av.z + hv.z * d2 + bv.z);
    o.w = tanhf(av.w + hv.w * d2 + bv.w);
    *(float4*)(z + (size_t)i * LDZ + coloff + f) = o;
}

// H=1 path (last GCN layer, column 96)
__global__ void gcn_combine1(const float* __restrict__ agg,
                             const float* __restrict__ hw,
                             const float* __restrict__ dis,
                             const float* __restrict__ bias,
                             float* __restrict__ z, int coloff) {
    int i = blockIdx.x * blockDim.x + threadIdx.x;
    if (i >= NN) return;
    float d = dis[i];
    z[(size_t)i * LDZ + coloff] = tanhf(agg[i] + hw[i] * d * d + bias[0]);
}

__global__ void count_nodes(const long long* __restrict__ batch, int* counts) {
    int i = blockIdx.x * blockDim.x + threadIdx.x;
    if (i < NN) atomicAdd(&counts[(int)batch[i]], 1);
}

__global__ void scan_starts(const int* __restrict__ counts, int* starts) {
    if (blockIdx.x == 0 && threadIdx.x == 0) {
        int s = 0;
        for (int b = 0; b < BB; ++b) { starts[b] = s; s += counts[b]; }
    }
}

// Rank-based stable sort-pool: node i's rank within its (contiguous) graph
// under key (-z[:,96], original index).  rank<K rows copied to zp[b][rank].
// Full padded 128-float row is copied (pad cols of z are zero -> correct).
__global__ void sort_pool(const float* __restrict__ z,
                          const long long* __restrict__ batch,
                          const int* __restrict__ starts,
                          const int* __restrict__ counts,
                          float* __restrict__ zp) {
    int i = blockIdx.x * blockDim.x + threadIdx.x;
    if (i >= NN) return;
    int b = (int)batch[i];
    int s = starts[b], c = counts[b];
    float vi = z[(size_t)i * LDZ + (DLAT - 1)];
    int rank = 0;
    for (int j = s; j < s + c; ++j) {
        float vj = z[(size_t)j * LDZ + (DLAT - 1)];
        rank += (vj > vi) || (vj == vi && j < i);
    }
    if (rank < KK) {
        float4* dst = (float4*)(zp + ((size_t)b * KK + rank) * LDZ);
        const float4* src = (const float4*)(z + (size_t)i * LDZ);
        for (int d = 0; d < LDZ / 4; ++d) dst[d] = src[d];
    }
}

// relu(conv1 + bias) followed by MaxPool1d(2,2):
// g1 is [B*K, C1] (GEMM out); p1 is [B][C1][K/2]
__global__ void pool_epilogue(const float* __restrict__ g1,
                              const float* __restrict__ b1,
                              float* __restrict__ p1) {
    int t = blockIdx.x * blockDim.x + threadIdx.x;
    const int total = BB * C1 * (KK / 2);
    if (t >= total) return;
    int b = t / (C1 * (KK / 2));
    int r = t - b * (C1 * (KK / 2));
    int c = r / (KK / 2);
    int kp = r - c * (KK / 2);
    float y0 = fmaxf(g1[((size_t)b * KK + 2 * kp)     * C1 + c] + b1[c], 0.0f);
    float y1 = fmaxf(g1[((size_t)b * KK + 2 * kp + 1) * C1 + c] + b1[c], 0.0f);
    p1[((size_t)b * C1 + c) * (KK / 2) + kp] = fmaxf(y0, y1);
}

// conv2 (NCH, valid) + relu, flattened straight into [B, DENSE] (o*T2 + t)
__global__ void conv2_kernel(const float* __restrict__ p1,
                             const float* __restrict__ w2,
                             const float* __restrict__ b2,
                             float* __restrict__ y2) {
    int t = blockIdx.x * blockDim.x + threadIdx.x;
    const int total = BB * C2 * T2;
    if (t >= total) return;
    int b = t / (C2 * T2);
    int r = t - b * (C2 * T2);
    int o = r / T2;
    int tt = r - o * T2;
    float acc = b2[o];
#pragma unroll
    for (int i = 0; i < C1; ++i)
#pragma unroll
        for (int w = 0; w < KW2; ++w)
            acc += p1[((size_t)b * C1 + i) * (KK / 2) + tt + w] *
                   w2[((size_t)o * C1 + i) * KW2 + w];
    y2[(size_t)b * DENSE + o * T2 + tt] = fmaxf(acc, 0.0f);
}

__global__ void lin2_kernel(const float* __restrict__ l1,
                            const float* __restrict__ w,
                            const float* __restrict__ bias,
                            float* __restrict__ out) {
    int b = blockIdx.x * blockDim.x + threadIdx.x;
    if (b >= BB) return;
    float acc = bias[0];
    for (int j = 0; j < 128; ++j) acc += l1[(size_t)b * 128 + j] * w[j];
    out[b] = acc;
}

// ---------------- host-side launch helpers ----------------
static inline int cdiv(long long a, int b) { return (int)((a + b - 1) / b); }

// Pack B fragments, then run the WMMA GEMM (NT=2 when N is a multiple
// of 32 -> A fragment reuse across two N-tiles).
static inline void launch_gemm(const float* A, int lda,
                               const float* Bm, int ldb, int btrans,
                               _Float16* frag,
                               float* C, int ldc, const float* bias, int relu,
                               int M, int K, int Kpad, int N, hipStream_t s) {
    int chunks = Kpad / 32;
    int tm = M / 16, tn = (N + 15) / 16;
    pack_b_frag<<<cdiv((long long)tn * chunks * 32, 256), 256, 0, s>>>(
        Bm, ldb, btrans, frag, K, N, chunks, tn);
    if (N % 32 == 0) {
        int tilesNG = tn / 2;
        int waves = tm * tilesNG;
        wmma_gemm_f16f32<2><<<(waves + 7) / 8, 256, 0, s>>>(
            A, lda, frag, C, ldc, bias, relu, M, chunks, N, tilesNG, waves);
    } else {
        int waves = tm * tn;
        wmma_gemm_f16f32<1><<<(waves + 7) / 8, 256, 0, s>>>(
            A, lda, frag, C, ldc, bias, relu, M, chunks, N, tn, waves);
    }
}

extern "C" void kernel_launch(void* const* d_in, const int* in_sizes, int n_in,
                              void* d_out, int out_size, void* d_ws, size_t ws_size,
                              hipStream_t stream) {
    // ---------------- unpack inputs (handle tuple flattening either way) ---
    const float*     x     = (const float*)d_in[0];
    const long long* ei    = (const long long*)d_in[1];
    const long long* batch = (const long long*)d_in[2];
    const float *gW[4], *gB[4];
    const float *c1W, *c1b, *c2W, *c2b, *l1W, *l1b, *l2W, *l2b;
    if (n_in >= 19) {                      // tuples expanded to separate entries
        for (int i = 0; i < 4; ++i) {
            gW[i] = (const float*)d_in[3 + i];
            gB[i] = (const float*)d_in[7 + i];
        }
        c1W = (const float*)d_in[11]; c1b = (const float*)d_in[12];
        c2W = (const float*)d_in[13]; c2b = (const float*)d_in[14];
        l1W = (const float*)d_in[15]; l1b = (const float*)d_in[16];
        l2W = (const float*)d_in[17]; l2b = (const float*)d_in[18];
    } else {                               // tuples passed as one flat array
        const float* gw = (const float*)d_in[3];
        gW[0] = gw;
        gW[1] = gW[0] + FIN * HH;
        gW[2] = gW[1] + HH * HH;
        gW[3] = gW[2] + HH * HH;
        const float* gb = (const float*)d_in[4];
        gB[0] = gb; gB[1] = gb + HH; gB[2] = gb + 2 * HH; gB[3] = gb + 3 * HH;
        c1W = (const float*)d_in[5];  c1b = (const float*)d_in[6];
        c2W = (const float*)d_in[7];  c2b = (const float*)d_in[8];
        l1W = (const float*)d_in[9];  l1b = (const float*)d_in[10];
        l2W = (const float*)d_in[11]; l2b = (const float*)d_in[12];
    }
    float* out = (float*)d_out;

    // ---------------- workspace layout (~78 MB) ----------------
    float* ws   = (float*)d_ws;
    float* dis  = ws;                              // N
    float* z    = dis + NN;                        // N * LDZ  (padded latents)
    float* hw   = z + (size_t)NN * LDZ;            // N * 32   (per-layer XW)
    float* agg  = hw + (size_t)NN * HH;            // N * 32   (edge aggregate)
    int*   cnt  = (int*)(agg + (size_t)NN * HH);   // B
    int*   sts  = cnt + BB;                        // B
    _Float16* frag = (_Float16*)(sts + BB + 512);  // 64K halves (32B aligned)
    // after the GCN stage hw/agg are dead -> reuse:
    float* zp = hw;                                // B*K*LDZ (1.97M < 3.2M)
    float* g1 = hw + (size_t)BB * KK * LDZ;        // B*K*C1  (0.25M)
    float* p1 = g1 + (size_t)BB * KK * C1;         // B*C1*15 (0.12M)
    float* y2 = agg;                               // B*352
    float* l1 = agg + (size_t)BB * DENSE;          // B*128

    // ---------------- degree / normalization ----------------
    fill_f32<<<cdiv(NN, 256), 256, 0, stream>>>(dis, 1.0f, NN);
    deg_edges<<<cdiv(EE, 256), 256, 0, stream>>>(ei, dis);
    rsqrt_inplace<<<cdiv(NN, 256), 256, 0, stream>>>(dis);

    // zero the padded z once (cols 97..127 stay zero; real cols overwritten)
    fill_f32<<<cdiv((long long)NN * LDZ, 256), 256, 0, stream>>>(
        z, 0.0f, NN * LDZ);

    // ---------------- 4 GCN layers, outputs concatenated into z ----------
    for (int l = 0; l < 4; ++l) {
        int ho          = (l == 3) ? 1 : HH;
        int coloff      = l * HH;
        const float* A  = (l == 0) ? x : (z + (l - 1) * HH);
        int lda         = (l == 0) ? FIN : LDZ;
        int kdim        = (l == 0) ? FIN : HH;
        // hw = A @ W   (WMMA; K already a multiple of 32)
        launch_gemm(A, lda, gW[l], ho, /*btrans=*/0, frag, hw, ho,
                    nullptr, 0, NN, kdim, kdim, ho, stream);
        // agg = scatter_add over edges
        fill_f32<<<cdiv((long long)NN * ho, 256), 256, 0, stream>>>(
            agg, 0.0f, NN * ho);
        if (ho == HH) {
            edge_agg32<<<cdiv((long long)EE * 8, 256), 256, 0, stream>>>(
                ei, dis, hw, agg);
            gcn_combine32<<<cdiv((long long)NN * 8, 256), 256, 0, stream>>>(
                agg, hw, dis, gB[l], z, coloff);
        } else {
            edge_agg1<<<cdiv(EE, 256), 256, 0, stream>>>(ei, dis, hw, agg);
            gcn_combine1<<<cdiv(NN, 256), 256, 0, stream>>>(
                agg, hw, dis, gB[l], z, coloff);
        }
    }

    // ---------------- sort-pool -> zp [B, K, LDZ] ----------------
    fill_i32<<<cdiv(BB, 256), 256, 0, stream>>>(cnt, 0, BB);
    count_nodes<<<cdiv(NN, 256), 256, 0, stream>>>(batch, cnt);
    scan_starts<<<1, 32, 0, stream>>>(cnt, sts);
    fill_f32<<<cdiv((long long)BB * KK * LDZ, 256), 256, 0, stream>>>(
        zp, 0.0f, BB * KK * LDZ);
    sort_pool<<<cdiv(NN, 128), 128, 0, stream>>>(z, batch, sts, cnt, zp);

    // ---------------- conv1 as GEMM [B*K,128pad] x [97,16] (WMMA) --------
    launch_gemm(zp, LDZ, c1W, DLAT, /*btrans=*/1, frag, g1, C1,
                nullptr, 0, BB * KK, DLAT, LDZ, C1, stream);
    pool_epilogue<<<cdiv(BB * C1 * (KK / 2), 256), 256, 0, stream>>>(g1, c1b, p1);

    // ---------------- conv2 + relu -> y2 [B, 352] ----------------
    conv2_kernel<<<cdiv(BB * C2 * T2, 256), 256, 0, stream>>>(p1, c2W, c2b, y2);

    // ---------------- lin1 (WMMA, fused bias+relu) -> l1 [B,128] ---------
    launch_gemm(y2, DENSE, l1W, DENSE, /*btrans=*/1, frag, l1, 128,
                l1b, /*relu=*/1, BB, DENSE, DENSE, 128, stream);

    // ---------------- lin2 -> out [B,1] ----------------
    lin2_kernel<<<cdiv(BB, 128), 128, 0, stream>>>(l1, l2W, l2b, out);
}
